// SimpleHarmonicRNN_37804302139635
// MI455X (gfx1250) — compile-verified, matrix operand order checked by reading, not run on hardware
//
#include <hip/hip_runtime.h>

typedef unsigned int u32;
typedef unsigned long long u64;
typedef int v4i __attribute__((vector_size(16)));
typedef __attribute__((ext_vector_type(16))) _Float16 v16h;
typedef __attribute__((ext_vector_type(8)))  float    v8f;

#define T_STEPS 500
#define BATCH   128
#define IN_DIM  700
#define H_DIM   1024
#define OUT_DIM 20
#define KPAD    704          // IN_DIM padded to multiple of 32
#define DT_C    0.01f
#define THETA_C 1.0f
#define GAMMA_C 0.9f

#define LDS_STRIDE 20        // dwords per LDS tile row: 16 data + 4 pad (16B-aligned rows)
#define BH (BATCH * H_DIM)

// CDNA5 async global->LDS path (ASYNCcnt-tracked), guarded so compile can't regress.
#if __has_builtin(__builtin_amdgcn_global_load_async_to_lds_b128) && \
    __has_builtin(__builtin_amdgcn_s_wait_asynccnt)
#define HAVE_ASYNC_LDS 1
#else
#define HAVE_ASYNC_LDS 0
#endif

#if HAVE_ASYNC_LDS
// Builtin signature (from compiler diagnostic): param0 = int4 addrspace(1)*,
// param1 = int4 addrspace(3)*, then imm offset + cpol. Generic->AS conversion
// goes via integer (AS3 ptr is the 32-bit LDS offset = low half of a generic
// LDS pointer; AS1 pointer value == the 64-bit global address).
__device__ __forceinline__ void async_g2l_b128(const u32* g, u32* l) {
  __builtin_amdgcn_global_load_async_to_lds_b128(
      (__attribute__((address_space(1))) v4i*)(u64)g,
      (__attribute__((address_space(3))) v4i*)(u64)l, 0, 0);
}
#endif

// ---------------------------------------------------------------------------
// Fragment gather from LDS matching the CDNA5 16-bit A/B WMMA layout:
// lane(0..15)=row, lanes 16..31 same rows with +8 K offset; VGPR r holds K pair.
// With a 20-dword row stride the 8 dwords per lane are two aligned 16B runs.
// ---------------------------------------------------------------------------
__device__ __forceinline__ v16h frag_from_lds(const u32* base, int row, int h16) {
  union { uint4 q[2]; v16h h; } f;
  const u32* p = base + row * LDS_STRIDE;
  f.q[0] = *(const uint4*)(p + 4 * h16);
  f.q[1] = *(const uint4*)(p + 8 + 4 * h16);
  return f.h;
}

// ---------------------------------------------------------------------------
// One-time: pack W_hidden (f32, [(IN+H), H]) into two transposed f16 buffers:
//   wxT[n][k] k<KPAD (zero-padded past IN_DIM), whT[n][k] k<H  (dword = 2 f16)
// ---------------------------------------------------------------------------
__global__ void __launch_bounds__(256) hrf_convert_w(const float* __restrict__ Wh,
                                                     u32* __restrict__ wxT,
                                                     u32* __restrict__ whT) {
  const int WX_DW = H_DIM * (KPAD / 2);
  const int WH_DW = H_DIM * (H_DIM / 2);
  int d = blockIdx.x * 256 + threadIdx.x;
  if (d < WX_DW) {
    int n = d / (KPAD / 2), kd = d % (KPAD / 2), k0 = kd * 2;
    union { _Float16 hh[2]; u32 u; } pk;
    pk.hh[0] = (k0     < IN_DIM) ? (_Float16)Wh[(size_t)k0       * H_DIM + n] : (_Float16)0.f;
    pk.hh[1] = (k0 + 1 < IN_DIM) ? (_Float16)Wh[(size_t)(k0 + 1) * H_DIM + n] : (_Float16)0.f;
    wxT[d] = pk.u;
  } else if (d < WX_DW + WH_DW) {
    int dd = d - WX_DW;
    int n = dd / (H_DIM / 2), kd = dd % (H_DIM / 2), k0 = kd * 2;
    union { _Float16 hh[2]; u32 u; } pk;
    pk.hh[0] = (_Float16)Wh[(size_t)(IN_DIM + k0)     * H_DIM + n];
    pk.hh[1] = (_Float16)Wh[(size_t)(IN_DIM + k0 + 1) * H_DIM + n];
    whT[dd] = pk.u;
  }
}

__global__ void __launch_bounds__(256) hrf_zero(u32* __restrict__ p, int n) {
  int i = blockIdx.x * 256 + threadIdx.x;
  if (i < n) p[i] = 0u;
}

// ---------------------------------------------------------------------------
// Phase 1: cur_x[T*B, H] = x[T*B, IN] @ W_x  (f16 WMMA, f32 accumulate)
// Block tile 128(M) x 64(N); 8 waves as 4(M) x 2(N); wave tile 32x32 (2x2 WMMA)
// A tile needs f32->f16 conversion (VALU staging); B tile is a pure f16 copy
// so it rides the CDNA5 async global->LDS engine when available.
// ---------------------------------------------------------------------------
__global__ void __launch_bounds__(256) hrf_gemm_x(const float* __restrict__ x,
                                                  const u32* __restrict__ wxT,
                                                  float* __restrict__ curx) {
  __shared__ alignas(16) u32 AsU[128 * LDS_STRIDE];
  __shared__ alignas(16) u32 BsU[64 * LDS_STRIDE];
  const int tid  = threadIdx.x;
  const int lane = tid & 31;
  const int h16  = (lane >> 4) & 1;
  const int l16  = lane & 15;
  const int wv   = tid >> 5;
  const int wm   = wv & 3;            // wave position along M (0..3)
  const int wn   = wv >> 2;           // wave position along N (0..1)
  const long rowBase = (long)blockIdx.x * 128;  // over T*B = 64000
  const int  colBase = blockIdx.y * 64;         // over H = 1024

  v8f acc[2][2] = {};

  for (int kc = 0; kc < KPAD; kc += 32) {
#if HAVE_ASYNC_LDS
    {  // B tile: 64 rows x 16 dwords = 256 lanes x b128, direct global->LDS
      int n = tid >> 2, seg = tid & 3;
      async_g2l_b128(wxT + (size_t)(colBase + n) * (KPAD / 2) + (kc >> 1) + seg * 4,
                     BsU + n * LDS_STRIDE + seg * 4);
    }
#endif
    // stage A: 128 rows x 32 K of x, f32 -> packed f16 pairs
#pragma unroll
    for (int i = 0; i < 8; ++i) {
      int d = tid + i * 256;                  // 0..2047 dwords
      int r = d >> 4, kd = d & 15;
      int k0 = kc + kd * 2;
      union { _Float16 hh[2]; u32 u; } pk;
      if (k0 < IN_DIM) {
        const float2 xv = *(const float2*)(x + (rowBase + r) * IN_DIM + k0);
        pk.hh[0] = (_Float16)xv.x; pk.hh[1] = (_Float16)xv.y;
      } else {
        pk.u = 0u;
      }
      AsU[r * LDS_STRIDE + kd] = pk.u;
    }
#if HAVE_ASYNC_LDS
    __builtin_amdgcn_s_wait_asynccnt(0);
#else
#pragma unroll
    for (int i = 0; i < 4; ++i) {
      int d = tid + i * 256;                  // 0..1023 dwords
      int n = d >> 4, kd = d & 15;
      BsU[n * LDS_STRIDE + kd] = wxT[(size_t)(colBase + n) * (KPAD / 2) + (kc >> 1) + kd];
    }
#endif
    __syncthreads();

    v16h af[2], bf[2];
#pragma unroll
    for (int mi = 0; mi < 2; ++mi) af[mi] = frag_from_lds(AsU, wm * 32 + mi * 16 + l16, h16);
#pragma unroll
    for (int ni = 0; ni < 2; ++ni) bf[ni] = frag_from_lds(BsU, wn * 32 + ni * 16 + l16, h16);
#pragma unroll
    for (int mi = 0; mi < 2; ++mi)
#pragma unroll
      for (int ni = 0; ni < 2; ++ni)
        acc[mi][ni] = __builtin_amdgcn_wmma_f32_16x16x32_f16(
            false, af[mi], false, bf[ni], (short)0, acc[mi][ni], false, false);
    __syncthreads();
  }

  // C layout: VGPR r -> row (r + 8*half), lane%16 -> col
#pragma unroll
  for (int mi = 0; mi < 2; ++mi)
#pragma unroll
    for (int ni = 0; ni < 2; ++ni)
#pragma unroll
      for (int r = 0; r < 8; ++r) {
        long gm = rowBase + wm * 32 + mi * 16 + r + 8 * h16;
        int  gn = colBase + wn * 32 + ni * 16 + l16;
        curx[gm * H_DIM + gn] = acc[mi][ni][r];
      }
}

// ---------------------------------------------------------------------------
// Phase 2 step: rec = z_prev @ W_h (WMMA), then HRF dynamics on the fragments.
// Blocks 0..15 own a 128x64 (batch x hidden) patch of state; both LDS tiles
// are pure f16 copies -> async double-buffered staging overlaps the next
// K-chunk's global fetch with the current chunk's WMMA work.
// Block 16 performs the fused leaky-integrator readout of the PREVIOUS step
// (it only needs zin = z_{t-1}), halving total launch count of the scan.
// z ping-pongs between two buffers to avoid cross-block read/write races.
// ---------------------------------------------------------------------------
__global__ void __launch_bounds__(256) hrf_step(const u32* __restrict__ whT,
                                                const float* __restrict__ curx_t,
                                                const float* __restrict__ omega,
                                                const float* __restrict__ boff,
                                                float* __restrict__ u,
                                                float* __restrict__ v,
                                                float* __restrict__ adap,
                                                const _Float16* __restrict__ zin,
                                                _Float16* __restrict__ zout,
                                                float* __restrict__ nsp,
                                                const float* __restrict__ Wout,
                                                const float* __restrict__ tau,
                                                float* __restrict__ outu,
                                                float* __restrict__ outPrev) {
  const int tid = threadIdx.x;

  if (blockIdx.x == 16) {   // -------- fused readout of previous timestep ----
    if (outPrev != nullptr) {
      for (int p = tid; p < BATCH * OUT_DIM; p += 256) {
        int bi = p / OUT_DIM, o = p % OUT_DIM;
        const _Float16* zr = zin + (size_t)bi * H_DIM;
        float s = 0.f;
        for (int h = 0; h < H_DIM; ++h) s += (float)zr[h] * Wout[h * OUT_DIM + o];
        float alpha = __expf(-DT_C / fabsf(tau[o]));
        float on = alpha * outu[p] + (1.f - alpha) * s;
        outu[p] = on;
        outPrev[p] = on;
      }
    }
    return;
  }

  __shared__ alignas(16) u32 AsU[2][128 * LDS_STRIDE];
  __shared__ alignas(16) u32 BsU[2][64 * LDS_STRIDE];
  __shared__ float ssum[256];
  const int lane = tid & 31;
  const int h16  = (lane >> 4) & 1;
  const int l16  = lane & 15;
  const int wv   = tid >> 5;
  const int wm   = wv & 3;
  const int wn   = wv >> 2;
  const int colBase = blockIdx.x * 64;
  const u32* zinU = (const u32*)zin;      // z already f16-packed

  v8f acc[2][2] = {};

#if HAVE_ASYNC_LDS
  // per-chunk staging: A = 512 b128 chunks (2/thread), B = 256 b128 (1/thread)
  // -> 3 ASYNCcnt-tracked instructions per wave per chunk.
  auto stage = [&](int kcX, int buf) {
#pragma unroll
    for (int j = 0; j < 2; ++j) {
      int c = tid + j * 256;                // 0..511
      int r = c >> 2, seg = c & 3;
      async_g2l_b128(zinU + (size_t)r * (H_DIM / 2) + (kcX >> 1) + seg * 4,
                     &AsU[buf][r * LDS_STRIDE + seg * 4]);
    }
    int n = tid >> 2, seg = tid & 3;
    async_g2l_b128(whT + (size_t)(colBase + n) * (H_DIM / 2) + (kcX >> 1) + seg * 4,
                   &BsU[buf][n * LDS_STRIDE + seg * 4]);
  };

  stage(0, 0);                              // prologue: chunk 0 -> buffer 0
  for (int kc = 0; kc < H_DIM; kc += 32) {
    const int cur = (kc >> 5) & 1;
    if (kc + 32 < H_DIM) {
      __syncthreads();                      // everyone done reading buf cur^1
      stage(kc + 32, cur ^ 1);              // overlap next fetch with this WMMA
      __builtin_amdgcn_s_wait_asynccnt(3);  // chunk kc retired (in-order), next 3 in flight
    } else {
      __builtin_amdgcn_s_wait_asynccnt(0);
    }
    __syncthreads();                        // chunk kc visible to all waves

    v16h af[2], bf[2];
#pragma unroll
    for (int mi = 0; mi < 2; ++mi) af[mi] = frag_from_lds(AsU[cur], wm * 32 + mi * 16 + l16, h16);
#pragma unroll
    for (int ni = 0; ni < 2; ++ni) bf[ni] = frag_from_lds(BsU[cur], wn * 32 + ni * 16 + l16, h16);
#pragma unroll
    for (int mi = 0; mi < 2; ++mi)
#pragma unroll
      for (int ni = 0; ni < 2; ++ni)
        acc[mi][ni] = __builtin_amdgcn_wmma_f32_16x16x32_f16(
            false, af[mi], false, bf[ni], (short)0, acc[mi][ni], false, false);
  }
  __syncthreads();
#else
  for (int kc = 0; kc < H_DIM; kc += 32) {
#pragma unroll
    for (int i = 0; i < 8; ++i) {
      int d = tid + i * 256;
      int r = d >> 4, kd = d & 15;
      AsU[0][r * LDS_STRIDE + kd] = zinU[r * (H_DIM / 2) + (kc >> 1) + kd];
    }
#pragma unroll
    for (int i = 0; i < 4; ++i) {
      int d = tid + i * 256;
      int n = d >> 4, kd = d & 15;
      BsU[0][n * LDS_STRIDE + kd] = whT[(size_t)(colBase + n) * (H_DIM / 2) + (kc >> 1) + kd];
    }
    if (kc + 32 < H_DIM)
      __builtin_prefetch(zinU + (tid >> 4) * (H_DIM / 2) + ((kc + 32) >> 1), 0, 1);
    __syncthreads();

    v16h af[2], bf[2];
#pragma unroll
    for (int mi = 0; mi < 2; ++mi) af[mi] = frag_from_lds(AsU[0], wm * 32 + mi * 16 + l16, h16);
#pragma unroll
    for (int ni = 0; ni < 2; ++ni) bf[ni] = frag_from_lds(BsU[0], wn * 32 + ni * 16 + l16, h16);
#pragma unroll
    for (int mi = 0; mi < 2; ++mi)
#pragma unroll
      for (int ni = 0; ni < 2; ++ni)
        acc[mi][ni] = __builtin_amdgcn_wmma_f32_16x16x32_f16(
            false, af[mi], false, bf[ni], (short)0, acc[mi][ni], false, false);
    __syncthreads();
  }
#endif

  // ---- HRF dynamics directly on accumulator fragments ----
  float spikes = 0.f;
#pragma unroll
  for (int mi = 0; mi < 2; ++mi)
#pragma unroll
    for (int ni = 0; ni < 2; ++ni)
#pragma unroll
      for (int r = 0; r < 8; ++r) {
        int bm = wm * 32 + mi * 16 + r + 8 * h16;        // batch index
        int hn = colBase + wn * 32 + ni * 16 + l16;      // hidden index
        int idx = bm * H_DIM + hn;
        float cur = acc[mi][ni][r] + curx_t[idx];
        float om = fabsf(omega[hn]);
        float bd = fabsf(boff[hn]);
        float u0 = u[idx], v0 = v[idx], a0 = adap[idx];
        float u1 = u0 + DT_C * (cur - bd * u0 - om * v0);
        float v1 = v0 + DT_C * (om * u1 - bd * v0);
        float zf = (u1 - THETA_C - a0) > 0.f ? 1.f : 0.f;
        float a1 = GAMMA_C * a0 + zf;
        u[idx] = u1; v[idx] = v1; adap[idx] = a1;
        zout[idx] = (_Float16)zf;
        spikes += zf;
      }

  ssum[tid] = spikes;
  __syncthreads();
  for (int s = 128; s > 0; s >>= 1) {
    if (tid < s) ssum[tid] += ssum[tid + s];
    __syncthreads();
  }
  if (tid == 0) atomicAdd(nsp, ssum[0]);
}

// ---------------------------------------------------------------------------
// Standalone readout (used once, for the last timestep after the scan)
// ---------------------------------------------------------------------------
__global__ void __launch_bounds__(256) hrf_readout(const _Float16* __restrict__ z,
                                                   const float* __restrict__ Wout,
                                                   const float* __restrict__ tau,
                                                   float* __restrict__ outu,
                                                   float* __restrict__ outT) {
  int tid = blockIdx.x * 256 + threadIdx.x;
  if (tid >= BATCH * OUT_DIM) return;
  int bi = tid / OUT_DIM, o = tid % OUT_DIM;
  const _Float16* zr = z + (size_t)bi * H_DIM;
  float s = 0.f;
  for (int h = 0; h < H_DIM; ++h) s += (float)zr[h] * Wout[h * OUT_DIM + o];
  float alpha = __expf(-DT_C / fabsf(tau[o]));
  float on = alpha * outu[tid] + (1.f - alpha) * s;
  outu[tid] = on;
  outT[tid] = on;
}

// ---------------------------------------------------------------------------
// Assemble tail of d_out: z(f32), u, v, a, out_u, nsp
// ---------------------------------------------------------------------------
__global__ void __launch_bounds__(256) hrf_final(const _Float16* __restrict__ z,
                                                 const float* __restrict__ u,
                                                 const float* __restrict__ v,
                                                 const float* __restrict__ adap,
                                                 const float* __restrict__ outu,
                                                 const float* __restrict__ nsp,
                                                 float* __restrict__ dst) {
  int tid = blockIdx.x * 256 + threadIdx.x;
  if (tid < BH)                             dst[tid] = (float)z[tid];
  else if (tid < 2 * BH)                    dst[tid] = u[tid - BH];
  else if (tid < 3 * BH)                    dst[tid] = v[tid - 2 * BH];
  else if (tid < 4 * BH)                    dst[tid] = adap[tid - 3 * BH];
  else if (tid < 4 * BH + BATCH * OUT_DIM)  dst[tid] = outu[tid - 4 * BH];
  else if (tid == 4 * BH + BATCH * OUT_DIM) dst[tid] = *nsp;
}

// ---------------------------------------------------------------------------
extern "C" void kernel_launch(void* const* d_in, const int* in_sizes, int n_in,
                              void* d_out, int out_size, void* d_ws, size_t ws_size,
                              hipStream_t stream) {
  const float* x     = (const float*)d_in[0];
  const float* Wh    = (const float*)d_in[1];
  const float* omega = (const float*)d_in[2];
  const float* boff  = (const float*)d_in[3];
  const float* Wout  = (const float*)d_in[4];
  const float* tau   = (const float*)d_in[5];
  float* out = (float*)d_out;

  char* ws = (char*)d_ws;
  size_t off = 0;
  float* curx = (float*)(ws + off); off += (size_t)T_STEPS * BH * sizeof(float);
  float* u    = (float*)(ws + off); off += (size_t)BH * sizeof(float);   // contiguous state block begins here
  float* v    = (float*)(ws + off); off += (size_t)BH * sizeof(float);
  float* adap = (float*)(ws + off); off += (size_t)BH * sizeof(float);
  _Float16* z0 = (_Float16*)(ws + off); off += (size_t)BH * sizeof(_Float16);
  _Float16* z1 = (_Float16*)(ws + off); off += (size_t)BH * sizeof(_Float16);
  float* outu = (float*)(ws + off); off += (size_t)BATCH * OUT_DIM * sizeof(float);
  float* nsp  = (float*)(ws + off); off += 16;   // padded so wxT/whT stay 16B-aligned
  u32* wxT = (u32*)(ws + off); off += (size_t)H_DIM * (KPAD / 2) * sizeof(u32);
  u32* whT = (u32*)(ws + off);

  // zero the contiguous state block (u,v,a,z0,z1,outu,nsp+pad)
  const int zeroDw = (3 * BH * 4 + 2 * BH * 2 + BATCH * OUT_DIM * 4 + 16) / 4;
  hrf_zero<<<(zeroDw + 255) / 256, 256, 0, stream>>>((u32*)u, zeroDw);

  // pack weights to transposed f16 once
  const int convDw = H_DIM * (KPAD / 2) + H_DIM * (H_DIM / 2);
  hrf_convert_w<<<(convDw + 255) / 256, 256, 0, stream>>>(Wh, wxT, whT);

  // phase 1: time-invariant input GEMM for all T at once
  hrf_gemm_x<<<dim3((T_STEPS * BATCH) / 128, H_DIM / 64), 256, 0, stream>>>(x, wxT, curx);

  // phase 2: sequential scan (z ping-pong; block 16 of step t emits readout t-1)
  _Float16* zb[2] = { z0, z1 };
  for (int t = 0; t < T_STEPS; ++t) {
    _Float16* zr = zb[t & 1];
    _Float16* zw = zb[(t + 1) & 1];
    float* outPrev = (t == 0) ? nullptr : out + (size_t)(t - 1) * BATCH * OUT_DIM;
    hrf_step<<<H_DIM / 64 + 1, 256, 0, stream>>>(whT, curx + (size_t)t * BH,
                                                 omega, boff, u, v, adap, zr, zw, nsp,
                                                 Wout, tau, outu, outPrev);
  }
  // readout of the final timestep (z_499 lives in buffer 0 since T is even)
  hrf_readout<<<(BATCH * OUT_DIM + 255) / 256, 256, 0, stream>>>(
      z0, Wout, tau, outu, out + (size_t)(T_STEPS - 1) * BATCH * OUT_DIM);

  // tail of d_out
  const int finalN = 4 * BH + BATCH * OUT_DIM + 1;
  hrf_final<<<(finalN + 255) / 256, 256, 0, stream>>>(z0, u, v, adap, outu, nsp,
                                                      out + (size_t)T_STEPS * BATCH * OUT_DIM);
}
